// MolGCN_7241314861279
// MI455X (gfx1250) — matrix operand order repaired and checked by reading, not compiled, not run on hardware
//
#include <hip/hip_runtime.h>
#include <hip/hip_bf16.h>

typedef __attribute__((ext_vector_type(16))) __bf16 v16bf;
typedef __attribute__((ext_vector_type(8)))  __bf16 v8bf;
typedef __attribute__((ext_vector_type(8)))  float  v8f;

#define N_NODES 100000
#define N_EDGES 1600000
#define N_GRAPHS 1000
#define DH 128
#define M_TILES (N_NODES / 16)   // 6250, exact

// ---------------------------------------------------------------- utilities
__global__ void zero_f32_kernel(float* __restrict__ p, long long n) {
    long long gid = (long long)blockIdx.x * blockDim.x + threadIdx.x;
    if (gid < n) p[gid] = 0.0f;
}

__global__ void degree_kernel(const int* __restrict__ dst, float* __restrict__ deg, int n_edges) {
    int e = blockIdx.x * blockDim.x + threadIdx.x;
    if (e < n_edges) atomicAdd(&deg[dst[e]], 1.0f);
}

__global__ void dinv_kernel(float* __restrict__ deg, int n_nodes) {
    int i = blockIdx.x * blockDim.x + threadIdx.x;
    if (i < n_nodes) deg[i] = rsqrtf(deg[i] + 1.0f);   // self-loop adds 1
}

// -------------------------------------------------- embedding w/ max_norm=1
__global__ void embed_kernel(const int* __restrict__ x, const float* __restrict__ emb,
                             __bf16* __restrict__ hbf, int n_nodes) {
    int gid  = blockIdx.x * blockDim.x + threadIdx.x;
    int node = gid >> 5;
    int lane = gid & 31;
    if (node >= n_nodes) return;
    int id = x[node];
    const float4* row = (const float4*)(emb + (size_t)id * DH);
    float4 v = row[lane];
    float ss = v.x * v.x + v.y * v.y + v.z * v.z + v.w * v.w;
    #pragma unroll
    for (int off = 16; off > 0; off >>= 1) ss += __shfl_xor(ss, off, 32);
    float nrm   = sqrtf(ss);
    float scale = fminf(1.0f, 1.0f / fmaxf(nrm, 1e-7f));
    __bf16* out = hbf + (size_t)node * DH + lane * 4;
    out[0] = (__bf16)(v.x * scale);
    out[1] = (__bf16)(v.y * scale);
    out[2] = (__bf16)(v.z * scale);
    out[3] = (__bf16)(v.w * scale);
}

// ------------------------------------- repack W[128x128] f32 -> B fragments
// Linear frag index g = t*128 + kb*32 + lane  (t = N-tile, kb = K-step).
// Lane holds col n = t*16 + (lane&15), K = kb*32 + (lane>=16 ? 8 : 0) + i + (i>=8 ? 8 : 0).
__global__ void repack_w_kernel(const float* __restrict__ W, __bf16* __restrict__ Bp) {
    int gid = blockIdx.x * blockDim.x + threadIdx.x;
    if (gid >= 1024) return;
    int lane = gid & 31;
    int kb   = (gid >> 5) & 3;
    int t    = gid >> 7;
    int n    = t * 16 + (lane & 15);
    int k0   = kb * 32 + ((lane >> 4) << 3);
    __bf16* dst = Bp + (size_t)gid * 16;
    #pragma unroll
    for (int i = 0; i < 16; ++i) dst[i] = (__bf16)W[(size_t)(k0 + i + (i >= 8 ? 8 : 0)) * DH + n];
}

// ---------------------------------------------- WMMA GEMM: hw = hbf @ W  (f32 out)
// One wave computes a 16x128 output tile: 8 N-tiles x 4 K-steps = 32 v_wmma.
__global__ __launch_bounds__(256) void gemm_bf16_kernel(
    const __bf16* __restrict__ A, const __bf16* __restrict__ Bp,
    float* __restrict__ C, int m_tiles) {
    int lane = threadIdx.x & 31;
    int wave = threadIdx.x >> 5;
    int tile = blockIdx.x * 8 + wave;
    if (tile >= m_tiles) return;           // wave-uniform; EXEC stays all-1s

    int row   = tile * 16 + (lane & 15);
    int khalf = (lane >> 4) << 3;          // 0 for lanes 0-15, 8 for 16-31

    v8f acc[8];
    #pragma unroll
    for (int t = 0; t < 8; ++t) acc[t] = (v8f){0, 0, 0, 0, 0, 0, 0, 0};

    #pragma unroll
    for (int kb = 0; kb < 4; ++kb) {
        const __bf16* ap = A + (size_t)row * DH + kb * 32 + khalf;
        v8bf alo = *(const v8bf*)(ap);        // K = base .. base+7
        v8bf ahi = *(const v8bf*)(ap + 16);   // K = base+16 .. base+23
        v16bf a;
        #pragma unroll
        for (int i = 0; i < 8; ++i) { a[i] = alo[i]; a[8 + i] = ahi[i]; }
        #pragma unroll
        for (int t = 0; t < 8; ++t) {
            v16bf b = *(const v16bf*)(Bp + (size_t)((t * 4 + kb) * 32 + lane) * 16);
            acc[t] = __builtin_amdgcn_wmma_f32_16x16x32_bf16(
                false, a, false, b, (short)0, acc[t], false, false);
        }
    }

    int nbase = lane & 15;
    int mrow  = tile * 16 + ((lane >> 4) << 3);   // VGPR r -> row mrow + r
    #pragma unroll
    for (int t = 0; t < 8; ++t) {
        float* cp = C + (size_t)mrow * DH + t * 16 + nbase;
        #pragma unroll
        for (int r = 0; r < 8; ++r) cp[(size_t)r * DH] = acc[t][r];
    }
}

// ------------------------------------------- edge scatter: agg[dst] += hw[src]*coef
__global__ void edge_agg_kernel(const int* __restrict__ src, const int* __restrict__ dst,
                                const float* __restrict__ dinv, const float* __restrict__ hw,
                                float* __restrict__ agg, int n_edges) {
    long long gid = (long long)blockIdx.x * blockDim.x + threadIdx.x;
    int e    = (int)(gid >> 5);
    int lane = (int)(gid & 31);
    if (e >= n_edges) return;
    int s = src[e], d = dst[e];
    float coef = dinv[s] * dinv[d];
    float4 v = ((const float4*)(hw + (size_t)s * DH))[lane];
    float* ap = agg + (size_t)d * DH + lane * 4;
    atomicAdd(ap + 0, v.x * coef);
    atomicAdd(ap + 1, v.y * coef);
    atomicAdd(ap + 2, v.z * coef);
    atomicAdd(ap + 3, v.w * coef);
}

// -------------------- finalize: h = relu(agg + hw*dinv^2 + b); emit f32 + bf16
__global__ void finalize_kernel(float* __restrict__ agg, const float* __restrict__ hw,
                                const float* __restrict__ dinv, const float* __restrict__ bias,
                                __bf16* __restrict__ hbf, long long n) {
    long long gid = (long long)blockIdx.x * blockDim.x + threadIdx.x;
    if (gid >= n) return;
    int node = (int)(gid >> 7);
    int f    = (int)(gid & 127);
    float di = dinv[node];
    float v  = agg[gid] + hw[gid] * di * di + bias[f];
    v = fmaxf(v, 0.0f);
    agg[gid] = v;
    hbf[gid] = (__bf16)v;
}

// -------------------------- segment_max pooling (h >= 0 -> uint-bits atomicMax)
__global__ void pool_kernel(const float* __restrict__ h, const int* __restrict__ batch,
                            unsigned* __restrict__ gbits, long long n) {
    long long gid = (long long)blockIdx.x * blockDim.x + threadIdx.x;
    if (gid >= n) return;
    int node = (int)(gid >> 7);
    int f    = (int)(gid & 127);
    int b    = batch[node];
    atomicMax(gbits + (size_t)b * DH + f, __float_as_uint(h[gid]));
}

// ---------------------------------------------- out[g] = dot(g[g], Wf) + bf
__global__ void final_kernel(const float* __restrict__ g, const float* __restrict__ Wf,
                             const float* __restrict__ bfin, float* __restrict__ out,
                             int n_graphs) {
    int gid   = blockIdx.x * blockDim.x + threadIdx.x;
    int graph = gid >> 5;
    int lane  = gid & 31;
    if (graph >= n_graphs) return;
    float4 a = ((const float4*)(g + (size_t)graph * DH))[lane];
    float4 w = ((const float4*)Wf)[lane];
    float p = a.x * w.x + a.y * w.y + a.z * w.z + a.w * w.w;
    #pragma unroll
    for (int off = 16; off > 0; off >>= 1) p += __shfl_xor(p, off, 32);
    if (lane == 0) out[graph] = p + bfin[0];
}

// ============================================================== launch
extern "C" void kernel_launch(void* const* d_in, const int* in_sizes, int n_in,
                              void* d_out, int out_size, void* d_ws, size_t ws_size,
                              hipStream_t stream) {
    (void)in_sizes; (void)n_in; (void)out_size; (void)ws_size;

    const int*   x    = (const int*)d_in[0];
    const int*   ei   = (const int*)d_in[1];
    const int*   srcp = ei;
    const int*   dstp = ei + N_EDGES;
    const int*   bat  = (const int*)d_in[2];
    const float* emb  = (const float*)d_in[3];
    const float* Wl[3] = {(const float*)d_in[4], (const float*)d_in[6], (const float*)d_in[8]};
    const float* bl[3] = {(const float*)d_in[5], (const float*)d_in[7], (const float*)d_in[9]};
    const float* Wf   = (const float*)d_in[10];
    const float* bf   = (const float*)d_in[11];
    float* out = (float*)d_out;

    // workspace carve-up (256B aligned)
    char* ws = (char*)d_ws;
    size_t off = 0;
    auto carve = [&](size_t bytes) { char* p = ws + off; off += (bytes + 255) & ~(size_t)255; return p; };
    __bf16* hbf  = (__bf16*)carve((size_t)N_NODES * DH * sizeof(__bf16));   // 25.6 MB
    float*  hw   = (float*) carve((size_t)N_NODES * DH * sizeof(float));    // 51.2 MB
    float*  agg  = (float*) carve((size_t)N_NODES * DH * sizeof(float));    // 51.2 MB
    float*  dinv = (float*) carve((size_t)N_NODES * sizeof(float));         // 0.4 MB
    __bf16* Wb   = (__bf16*)carve((size_t)1024 * 16 * sizeof(__bf16));      // 32 KB
    float*  gbuf = (float*) carve((size_t)N_GRAPHS * DH * sizeof(float));   // 0.5 MB

    const int TB = 256;
    const long long NF = (long long)N_NODES * DH;      // 12.8M
    dim3 blkNF((unsigned)((NF + TB - 1) / TB));
    dim3 blkN((N_NODES + TB - 1) / TB);

    // degree -> dinv (in place)
    zero_f32_kernel<<<blkN, TB, 0, stream>>>(dinv, N_NODES);
    degree_kernel<<<(N_EDGES + TB - 1) / TB, TB, 0, stream>>>(dstp, dinv, N_EDGES);
    dinv_kernel<<<blkN, TB, 0, stream>>>(dinv, N_NODES);

    // embedding with max_norm renorm -> bf16 activations
    embed_kernel<<<(N_NODES * 32 + TB - 1) / TB, TB, 0, stream>>>(x, emb, hbf, N_NODES);

    // 3 GCN layers
    for (int L = 0; L < 3; ++L) {
        repack_w_kernel<<<4, TB, 0, stream>>>(Wl[L], Wb);
        gemm_bf16_kernel<<<(M_TILES + 7) / 8, TB, 0, stream>>>(hbf, Wb, hw, M_TILES);
        zero_f32_kernel<<<blkNF, TB, 0, stream>>>(agg, NF);
        edge_agg_kernel<<<(unsigned)(((long long)N_EDGES * 32 + TB - 1) / TB), TB, 0, stream>>>(
            srcp, dstp, dinv, hw, agg, N_EDGES);
        finalize_kernel<<<blkNF, TB, 0, stream>>>(agg, hw, dinv, bl[L], hbf, NF);
    }

    // graph max-pool + final linear
    zero_f32_kernel<<<((N_GRAPHS * DH) + TB - 1) / TB, TB, 0, stream>>>(gbuf, (long long)N_GRAPHS * DH);
    pool_kernel<<<blkNF, TB, 0, stream>>>(agg, bat, (unsigned*)gbuf, NF);
    final_kernel<<<(N_GRAPHS * 32 + TB - 1) / TB, TB, 0, stream>>>(gbuf, Wf, bf, out, N_GRAPHS);
}
